// prototypeLayer_Bert_50818053047015
// MI455X (gfx1250) — matrix operand order repaired and checked by reading, not compiled, not run on hardware
//
#include <hip/hip_runtime.h>
#include <stdint.h>

// ---------------------------------------------------------------------------
// dist[b,s,p] = ||x||^2 + ||p||^2 - 2 x.p      X:[16384,768]f32  P:[512,768]f32
// out = dist [16384,512] f32  ++  prototype copy [512,768] f32
//
// Pass 1: split fp32 -> bf16 (hi, lo) arrays in d_ws (L2-resident working set),
//         row sums of squares, prototype copy (NT) to the output tail.
// Pass 2: 128x128-tile WMMA GEMM; K staged 32-wide through double-buffered LDS
//         via GLOBAL_LOAD_ASYNC_TO_LDS_B128 (ASYNCcnt) software pipeline;
//         cross = Ah*Bh + Ah*Bl + Al*Bh on v_wmma_f32_16x16x32_bf16
//         (per-product error ~2^-17, fp32 accumulate); fused distance epilogue
//         streamed out with non-temporal stores.
// ---------------------------------------------------------------------------

typedef __attribute__((ext_vector_type(16))) __bf16 v16bf;
typedef __attribute__((ext_vector_type(8)))  __bf16 v8bf;
typedef __attribute__((ext_vector_type(4)))  __bf16 v4bf;
typedef __attribute__((ext_vector_type(8)))  float  v8f;
typedef __attribute__((ext_vector_type(4)))  float  v4f;   // clang vector: OK for NT builtins

constexpr int kB = 32, kS = 512, kP = 512, kD = 768;
constexpr int kM = kB * kS;   // 16384
constexpr int kN = kP;        // 512
constexpr int kK = kD;        // 768

constexpr int TM = 128, TN = 128, TK = 32;
constexpr int LSTR = 40;          // LDS row stride (bf16): 80 B rows, conflict-free
constexpr int KSTEPS = kK / TK;   // 24

// ---- CDNA5 async global->LDS copy (ASYNCcnt) ------------------------------
__device__ __forceinline__ void async_copy_b128(uint32_t lds_off, const void* g)
{
    // VDST = LDS byte address (per-lane VGPR), VADDR = 64-bit global address.
    asm volatile("global_load_async_to_lds_b128 %0, %1, off"
                 :: "v"(lds_off), "v"(g) : "memory");
}
__device__ __forceinline__ void wait_async_all()
{
    asm volatile("s_wait_asynccnt 0x0" ::: "memory");
}

// ---------------------------------------------------------------------------
// Pass 1: one wave32 per row: bf16 hi/lo split, row sum-of-squares,
// optional fp32 row copy (prototypes -> output tail, non-temporal).
// ---------------------------------------------------------------------------
__global__ __launch_bounds__(256) void split_rows_kernel(
    const float* __restrict__ src,
    __bf16* __restrict__ hi, __bf16* __restrict__ lo,
    float* __restrict__ sq, float* __restrict__ copy_out)
{
    const int row  = blockIdx.x * 8 + (threadIdx.x >> 5);
    const int lane = threadIdx.x & 31;
    const size_t rb = (size_t)row * kK;

    float s = 0.f;
#pragma unroll
    for (int i = 0; i < kK / (32 * 4); ++i) {         // 6 vec4 per lane
        const int c = (lane + 32 * i) * 4;
        const v4f v = __builtin_nontemporal_load((const v4f*)&src[rb + c]);
        if (copy_out) __builtin_nontemporal_store(v, (v4f*)&copy_out[rb + c]);
        v4bf h4, l4;
#pragma unroll
        for (int j = 0; j < 4; ++j) {
            const float f = v[j];
            __bf16 h = (__bf16)f;
            h4[j] = h;
            l4[j] = (__bf16)(f - (float)h);
            s += f * f;
        }
        *(v4bf*)&hi[rb + c] = h4;     // regular-temporal: keep L2-resident
        *(v4bf*)&lo[rb + c] = l4;
    }
#pragma unroll
    for (int off = 16; off > 0; off >>= 1)
        s += __shfl_xor(s, off, 32);
    if (lane == 0) sq[row] = s;
}

// ---------------------------------------------------------------------------
// Pass 2: WMMA GEMM with async double-buffered LDS staging.
// 8 wave32s arranged 4(M) x 2(N); each wave: 2x4 v8f accumulators (32x64).
// ---------------------------------------------------------------------------
__global__ __launch_bounds__(256) void dist_wmma_async_kernel(
    const __bf16* __restrict__ Xh, const __bf16* __restrict__ Xl,
    const __bf16* __restrict__ Ph, const __bf16* __restrict__ Pl,
    const float* __restrict__ xsq, const float* __restrict__ psq,
    float* __restrict__ out)
{
    // [buffer][array: Ah,Al,Bh,Bl][row][k]
    __shared__ __align__(16) __bf16 lds[2][4][TM][LSTR];

    const int tid  = threadIdx.x;
    const int lane = tid & 31;
    const int wave = tid >> 5;
    const int wm   = wave & 3;
    const int wn   = wave >> 2;
    const int rowBase = blockIdx.y * TM;
    const int colBase = blockIdx.x * TN;

    const __bf16* gb[4] = { Xh + (size_t)rowBase * kK, Xl + (size_t)rowBase * kK,
                            Ph + (size_t)colBase * kK, Pl + (size_t)colBase * kK };

    const int chunk = tid & 3;     // 16B chunk within a 64B (32-elem) tile row
    const int row0  = tid >> 2;    // 0..63

    auto stage = [&](int buf, int k0) {
#pragma unroll
        for (int sel = 0; sel < 4; ++sel) {
#pragma unroll
            for (int h = 0; h < 2; ++h) {
                const int row = row0 + 64 * h;
                const __bf16* g = gb[sel] + (size_t)row * kK + k0 + chunk * 8;
                const uint32_t l =
                    (uint32_t)(uintptr_t)&lds[buf][sel][row][chunk * 8];
                async_copy_b128(l, g);
            }
        }
    };

    const v8f vzero = {0.f, 0.f, 0.f, 0.f, 0.f, 0.f, 0.f, 0.f};
    v8f acc[2][4];
#pragma unroll
    for (int a = 0; a < 2; ++a)
#pragma unroll
        for (int b = 0; b < 4; ++b) acc[a][b] = vzero;

    stage(0, 0);   // prologue: async-fill buffer 0

    const int mrow = lane & 15;
    const int kbA  = (lane & 16) ? 8 : 0;
    const int kbB  = (lane & 16) ? 8 : 0;

    for (int ks = 0; ks < KSTEPS; ++ks) {
        wait_async_all();        // my stage(ks) copies landed in LDS
        __syncthreads();         // all landed; all waves done with prev buffer
        if (ks + 1 < KSTEPS) stage((ks + 1) & 1, (ks + 1) * TK);
        const int cur = ks & 1;

        // ---- fragments (ISA 16-bit layouts, wave32).
        v16bf Af_h[2], Af_l[2];
#pragma unroll
        for (int a = 0; a < 2; ++a) {
            const int m = wm * 32 + a * 16 + mrow;
            v8bf c1h = *(const v8bf*)&lds[cur][0][m][kbA];
            v8bf c2h = *(const v8bf*)&lds[cur][0][m][kbA + 16];
            v8bf c1l = *(const v8bf*)&lds[cur][1][m][kbA];
            v8bf c2l = *(const v8bf*)&lds[cur][1][m][kbA + 16];
            Af_h[a] = __builtin_shufflevector(c1h, c2h, 0,1,2,3,4,5,6,7,8,9,10,11,12,13,14,15);
            Af_l[a] = __builtin_shufflevector(c1l, c2l, 0,1,2,3,4,5,6,7,8,9,10,11,12,13,14,15);
        }
        v16bf Bf_h[4], Bf_l[4];
#pragma unroll
        for (int b = 0; b < 4; ++b) {
            const int n = wn * 64 + b * 16 + mrow;
            v8bf c1h = *(const v8bf*)&lds[cur][2][n][kbB];
            v8bf c2h = *(const v8bf*)&lds[cur][2][n][kbB + 16];
            v8bf c1l = *(const v8bf*)&lds[cur][3][n][kbB];
            v8bf c2l = *(const v8bf*)&lds[cur][3][n][kbB + 16];
            Bf_h[b] = __builtin_shufflevector(c1h, c2h, 0,1,2,3,4,5,6,7,8,9,10,11,12,13,14,15);
            Bf_l[b] = __builtin_shufflevector(c1l, c2l, 0,1,2,3,4,5,6,7,8,9,10,11,12,13,14,15);
        }

        // ---- split-precision cross term: hh + hl + lh, fp32 accumulate.
#pragma unroll
        for (int a = 0; a < 2; ++a)
#pragma unroll
            for (int b = 0; b < 4; ++b) {
                acc[a][b] = __builtin_amdgcn_wmma_f32_16x16x32_bf16(
                    false, Af_h[a], false, Bf_h[b], (short)0, acc[a][b], false, false);
                acc[a][b] = __builtin_amdgcn_wmma_f32_16x16x32_bf16(
                    false, Af_h[a], false, Bf_l[b], (short)0, acc[a][b], false, false);
                acc[a][b] = __builtin_amdgcn_wmma_f32_16x16x32_bf16(
                    false, Af_l[a], false, Bf_h[b], (short)0, acc[a][b], false, false);
            }
    }

    // ---- epilogue: dist = x_sq[m] + p_sq[n] - 2*cross  (C/D fragment map),
    // streamed with non-temporal stores (output is never re-read).
    const int mhalf = (lane & 16) ? 8 : 0;
#pragma unroll
    for (int b = 0; b < 4; ++b) {
        const int n  = colBase + wn * 64 + b * 16 + (lane & 15);
        const float pn = psq[n];
#pragma unroll
        for (int a = 0; a < 2; ++a) {
            const int mBase = rowBase + wm * 32 + a * 16 + mhalf;
#pragma unroll
            for (int r = 0; r < 8; ++r) {
                const int m = mBase + r;
                const float d = xsq[m] + pn - 2.0f * acc[a][b][r];
                __builtin_nontemporal_store(d, &out[(size_t)m * kN + n]);
            }
        }
    }
}

// ---------------------------------------------------------------------------
extern "C" void kernel_launch(void* const* d_in, const int* in_sizes, int n_in,
                              void* d_out, int out_size, void* d_ws, size_t ws_size,
                              hipStream_t stream)
{
    (void)in_sizes; (void)n_in; (void)out_size; (void)ws_size;
    const float* X  = (const float*)d_in[0];
    const float* Pr = (const float*)d_in[1];
    float* out       = (float*)d_out;
    float* proto_out = out + (size_t)kM * kN;

    // workspace layout (needs ~52.5 MB; fits the 192 MB L2 as working set)
    char* ws = (char*)d_ws;
    __bf16* Xh  = (__bf16*)ws;
    __bf16* Xl  = (__bf16*)(ws + (size_t)kM * kK * 2);
    __bf16* Phh = (__bf16*)(ws + (size_t)kM * kK * 4);
    __bf16* Pll = (__bf16*)(ws + (size_t)kM * kK * 4 + (size_t)kN * kK * 2);
    float*  xsq = (float*) (ws + (size_t)kM * kK * 4 + (size_t)kN * kK * 4);
    float*  psq = xsq + kM;

    split_rows_kernel<<<kM / 8, 256, 0, stream>>>(X,  Xh,  Xl,  xsq, nullptr);
    split_rows_kernel<<<kN / 8, 256, 0, stream>>>(Pr, Phh, Pll, psq, proto_out);

    dim3 grid(kN / TN, kM / TM);   // (4, 128)
    dist_wmma_async_kernel<<<grid, 256, 0, stream>>>(Xh, Xl, Phh, Pll, xsq, psq, out);
}
